// MixtureOfExperts_2302102471008
// MI455X (gfx1250) — matrix operand order repaired and checked by reading, not compiled, not run on hardware
//
#include <hip/hip_runtime.h>
#include <stdint.h>
#include <math.h>

// Problem sizes
#define B_TOK 2048
#define DIM   1024
#define HID   4096
#define NE    8
#define NSLOT (B_TOK * 2)   // B * TOP_K routed slots

// d_out float offsets (output, gates, top_i, loss, utilization)
#define G_OFF    (B_TOK * DIM)
#define TI_OFF   (G_OFF + B_TOK * NE)
#define LBL_OFF  (TI_OFF + B_TOK * 2)
#define UTIL_OFF (LBL_OFF + 1)
#define OUT_TOTAL (UTIL_OFF + NE)

// workspace byte offsets
#define OFF_CNT    0
#define OFF_LISTS  256
#define OFF_SLOTG  (OFF_LISTS + NE * B_TOK * 4)
#define OFF_PROBS  (OFF_SLOTG + NSLOT * 4)
#define OFF_IMP    (OFF_PROBS + B_TOK * NE * 4)
#define OFF_LOAD   (OFF_IMP + 32)
#define OFF_UTIL   (OFF_LOAD + 32)
#define OFF_XBF    (((OFF_UTIL + 32) + 255) & ~255)
#define OFF_HBUF   (OFF_XBF + B_TOK * DIM * 2)

// GEMM tiling
#define TM 32      // rows (routed slots) per workgroup
#define TN 256     // output cols per workgroup
#define TK 32      // K per WMMA step (bf16)
#define LDP 40     // padded LDS row length in halves

typedef __attribute__((ext_vector_type(16))) __bf16 v16bf;
typedef __attribute__((ext_vector_type(8)))  float  v8f;
typedef __attribute__((ext_vector_type(4)))  unsigned int u32x4;
typedef __attribute__((ext_vector_type(4)))  int          i32x4;
typedef __attribute__((ext_vector_type(8)))  int          i32x8;

union FragBF { v16bf v; uint2 q[4]; unsigned short u[16]; };

__device__ __forceinline__ unsigned short f2bf(float f) {
  union { float f; uint32_t u; } c; c.f = f;
  return (unsigned short)((c.u + 0x7FFFu + ((c.u >> 16) & 1u)) >> 16);
}

// ---------------- Tensor Data Mover: async DMA of an f32 tile into LDS ----------------
// 2D tile: tile_h rows x tile_w f32 elements, row pitch = row_stride elements,
// packed contiguously at LDS byte offset lds_addr. Tracked by TENSORcnt.
__device__ __forceinline__ void tdm_load_tile_f32(
    uint32_t lds_addr, const float* gptr,
    uint32_t tile_w, uint32_t tile_h,
    uint32_t row_stride, uint32_t tensor_w, uint32_t tensor_h)
{
  uint64_t ga = (uint64_t)(uintptr_t)gptr;
  u32x4 g0;
  g0[0] = 1u;                                              // count=1 (valid user D#)
  g0[1] = lds_addr;                                        // lds_addr (bytes)
  g0[2] = (uint32_t)(ga & 0xFFFFFFFFu);                    // global_addr[31:0]
  g0[3] = (uint32_t)((ga >> 32) & 0x1FFFFFFu) | (2u << 30);// global_addr[56:32] | type=2
  i32x8 g1;
  g1[0] = (int)(2u << 16);                                 // wg_mask=0, data_size=2 (4B)
  g1[1] = (int)((tensor_w & 0xFFFFu) << 16);               // tensor_dim0[15:0]
  g1[2] = (int)(((tensor_w >> 16) & 0xFFFFu) |
                ((tensor_h & 0xFFFFu) << 16));             // tensor_dim0[31:16] | tensor_dim1[15:0]
  g1[3] = (int)(((tensor_h >> 16) & 0xFFFFu) |
                ((tile_w & 0xFFFFu) << 16));               // tensor_dim1[31:16] | tile_dim0
  g1[4] = (int)(tile_h & 0xFFFFu);                         // tile_dim1 (tile_dim2=0)
  g1[5] = (int)row_stride;                                 // tensor_dim0_stride[31:0]
  g1[6] = 0;                                               // stride[47:32], dim1_stride[15:0]
  g1[7] = 0;
  i32x4 z4 = {0, 0, 0, 0};
#if defined(__clang_major__) && (__clang_major__ >= 23)
  i32x8 z8 = {0, 0, 0, 0, 0, 0, 0, 0};
  __builtin_amdgcn_tensor_load_to_lds(g0, g1, z4, z4, z8, 0);
#else
  __builtin_amdgcn_tensor_load_to_lds(g0, g1, z4, z4, 0);
#endif
}

// ---------------- init ----------------
__global__ void zero_kernel(float* __restrict__ dout, int* __restrict__ cnt) {
  size_t i = (size_t)blockIdx.x * blockDim.x + threadIdx.x;
  size_t stride = (size_t)gridDim.x * blockDim.x;
  for (size_t k = i; k < (size_t)OUT_TOTAL; k += stride) dout[k] = 0.0f;
  if (i < NE) cnt[i] = 0;
}

__global__ void xcast_kernel(const float* __restrict__ x, unsigned short* __restrict__ xbf) {
  size_t i = ((size_t)blockIdx.x * blockDim.x + threadIdx.x) * 4;
  if (i >= (size_t)B_TOK * DIM) return;
  float4 f = *(const float4*)(x + i);
  unsigned short o[4] = { f2bf(f.x), f2bf(f.y), f2bf(f.z), f2bf(f.w) };
  *(uint2*)(xbf + i) = *(const uint2*)o;
}

// ---------------- gating: one wave (32 lanes) per token ----------------
__global__ __launch_bounds__(256) void gate_kernel(
    const float* __restrict__ x, const float* __restrict__ Wg,
    const float* __restrict__ bg, float* __restrict__ dout,
    float* __restrict__ probs, int* __restrict__ cnt,
    int* __restrict__ lists, float* __restrict__ slotg)
{
  int tok  = (int)((blockIdx.x * blockDim.x + threadIdx.x) >> 5);
  int lane = threadIdx.x & 31;
  if (tok >= B_TOK) return;
  const float* xr = x + (size_t)tok * DIM;

  float acc[NE];
  #pragma unroll
  for (int e = 0; e < NE; ++e) acc[e] = 0.0f;
  for (int d = lane; d < DIM; d += 32) {
    float xv = xr[d];
    const float* wr = Wg + (size_t)d * NE;
    #pragma unroll
    for (int e = 0; e < NE; ++e) acc[e] = fmaf(xv, wr[e], acc[e]);
  }
  #pragma unroll
  for (int e = 0; e < NE; ++e) {
    #pragma unroll
    for (int off = 16; off > 0; off >>= 1) acc[e] += __shfl_xor(acc[e], off, 32);
  }

  if (lane == 0) {
    float lg[NE];
    #pragma unroll
    for (int e = 0; e < NE; ++e) lg[e] = acc[e] + bg[e];
    int i0 = 0;
    #pragma unroll
    for (int e = 1; e < NE; ++e) if (lg[e] > lg[i0]) i0 = e;
    int i1 = (i0 == 0) ? 1 : 0;
    #pragma unroll
    for (int e = 0; e < NE; ++e) if (e != i1 && e != i0 && lg[e] > lg[i1]) i1 = e;
    float m2 = fmaxf(lg[i0], lg[i1]);
    float e0 = __expf(lg[i0] - m2), e1 = __expf(lg[i1] - m2);
    float inv = 1.0f / (e0 + e1);
    float g0 = e0 * inv, g1 = e1 * inv;
    float m8 = lg[0];
    #pragma unroll
    for (int e = 1; e < NE; ++e) m8 = fmaxf(m8, lg[e]);
    float s = 0.0f, p[NE];
    #pragma unroll
    for (int e = 0; e < NE; ++e) { p[e] = __expf(lg[e] - m8); s += p[e]; }
    float is = 1.0f / s;
    #pragma unroll
    for (int e = 0; e < NE; ++e) probs[tok * NE + e] = p[e] * is;

    dout[G_OFF + tok * NE + i0] = g0;
    dout[G_OFF + tok * NE + i1] = g1;
    dout[TI_OFF + tok * 2 + 0] = (float)i0;
    dout[TI_OFF + tok * 2 + 1] = (float)i1;

    int r0 = tok * 2, r1 = tok * 2 + 1;
    slotg[r0] = g0; slotg[r1] = g1;
    int p0 = atomicAdd(&cnt[i0], 1); lists[i0 * B_TOK + p0] = r0;
    int p1 = atomicAdd(&cnt[i1], 1); lists[i1 * B_TOK + p1] = r1;
  }
}

// ---------------- FFN1: h = gelu(x @ w1[e] + b1[e]), routed, TDM + bf16 WMMA ----------------
__global__ __launch_bounds__(256) void ffn1_kernel(
    const unsigned short* __restrict__ xbf,   // [B, DIM] bf16
    const float* __restrict__ w1,             // [E, DIM, HID]
    const float* __restrict__ b1,             // [E, HID]
    unsigned short* __restrict__ hbuf,        // [NSLOT, HID] bf16
    const int* __restrict__ cnt, const int* __restrict__ lists)
{
  __shared__ float stage[TK][TN];             // raw f32 weight tile (TDM target)
  __shared__ unsigned short Al[TM][LDP];
  __shared__ unsigned short Bt[TN][LDP];
  __shared__ int rowSlot[TM];

  const int e = blockIdx.z;
  const int n = cnt[e];
  const int tileBase = blockIdx.y * TM;
  if (tileBase >= n) return;
  const int hBase = blockIdx.x * TN;
  const int t = threadIdx.x;

  if (t < TM) {
    int idx = tileBase + t;
    rowSlot[t] = (idx < n) ? lists[e * B_TOK + idx] : lists[e * B_TOK];
  }
  __syncthreads();

  const int lane = t & 31, w = t >> 5;
  const int waveM = w & 1, waveN = w >> 1;
  const int half = lane >> 4, ln = lane & 15;

  v8f zero = {0.f,0.f,0.f,0.f,0.f,0.f,0.f,0.f};
  v8f acc[4] = { zero, zero, zero, zero };

  const float* wptr = w1 + (size_t)e * DIM * HID + hBase;
  const uint32_t stage_lds = (uint32_t)(uintptr_t)&stage[0][0];
  const int arow = t >> 3, ac4 = (t & 7) * 4;   // A fill: 4 halves/thread
  const size_t aSrc = (size_t)(rowSlot[arow] >> 1) * DIM;
  const int bk = t & 31, bc0 = (t >> 5) * 32;   // convert: 32 elements/thread

  if (w == 0)  // wave-level TDM issue for K-tile 0
    tdm_load_tile_f32(stage_lds, wptr, TN, TK, HID, HID, DIM);

  for (int kk = 0; kk < DIM / TK; ++kk) {
    if (w == 0) __builtin_amdgcn_s_wait_tensorcnt(0);
    __syncthreads();                          // stage(kk) ready; Bt free

    *(uint2*)&Al[arow][ac4] = *(const uint2*)&xbf[aSrc + kk * TK + ac4];
    #pragma unroll
    for (int i = 0; i < 8; ++i) {
      float4 f = *(const float4*)&stage[bk][bc0 + i * 4];
      Bt[bc0 + i*4 + 0][bk] = f2bf(f.x);
      Bt[bc0 + i*4 + 1][bk] = f2bf(f.y);
      Bt[bc0 + i*4 + 2][bk] = f2bf(f.z);
      Bt[bc0 + i*4 + 3][bk] = f2bf(f.w);
    }
    __syncthreads();                          // Bt ready; stage free

    if (w == 0 && kk + 1 < DIM / TK)          // DMA next tile, overlapped with WMMA
      tdm_load_tile_f32(stage_lds, wptr + (size_t)(kk + 1) * TK * HID, TN, TK, HID, HID, DIM);

    FragBF a;
    const int ar = waveM * 16 + ln, ko = half * 8;
    a.q[0] = *(const uint2*)&Al[ar][ko];
    a.q[1] = *(const uint2*)&Al[ar][ko + 4];
    a.q[2] = *(const uint2*)&Al[ar][ko + 16];
    a.q[3] = *(const uint2*)&Al[ar][ko + 20];
    const int kob = half * 8;
    #pragma unroll
    for (int j = 0; j < 4; ++j) {
      FragBF b;
      const int col = waveN * 64 + j * 16 + ln;
      b.q[0] = *(const uint2*)&Bt[col][kob];
      b.q[1] = *(const uint2*)&Bt[col][kob + 4];
      b.q[2] = *(const uint2*)&Bt[col][kob + 16];
      b.q[3] = *(const uint2*)&Bt[col][kob + 20];
      acc[j] = __builtin_amdgcn_wmma_f32_16x16x32_bf16(false, a.v, false, b.v,
                                                       (short)0, acc[j], false, false);
    }
  }

  #pragma unroll
  for (int j = 0; j < 4; ++j) {
    const int col = hBase + waveN * 64 + j * 16 + ln;
    const float bias = b1[e * HID + col];
    #pragma unroll
    for (int v = 0; v < 8; ++v) {
      const int mrow = waveM * 16 + half * 8 + v;
      if (tileBase + mrow < n) {
        const int r = rowSlot[mrow];
        float val = acc[j][v] + bias;
        val = 0.5f * val * (1.0f + erff(val * 0.70710678118654752f));  // exact GELU
        hbuf[(size_t)r * HID + col] = f2bf(val);
      }
    }
  }
}

// ---------------- FFN2: out += gate * (h @ w2[e] + b2[e]), routed, TDM + WMMA ----------------
__global__ __launch_bounds__(256) void ffn2_kernel(
    const unsigned short* __restrict__ hbuf,  // [NSLOT, HID] bf16
    const float* __restrict__ w2,             // [E, HID, DIM]
    const float* __restrict__ b2,             // [E, DIM]
    float* __restrict__ outp,                 // [B, DIM]
    const float* __restrict__ slotg,
    const int* __restrict__ cnt, const int* __restrict__ lists)
{
  __shared__ float stage[TK][TN];
  __shared__ unsigned short Al[TM][LDP];
  __shared__ unsigned short Bt[TN][LDP];
  __shared__ int rowSlot[TM];

  const int e = blockIdx.z;
  const int n = cnt[e];
  const int tileBase = blockIdx.y * TM;
  if (tileBase >= n) return;
  const int dBase = blockIdx.x * TN;
  const int t = threadIdx.x;

  if (t < TM) {
    int idx = tileBase + t;
    rowSlot[t] = (idx < n) ? lists[e * B_TOK + idx] : lists[e * B_TOK];
  }
  __syncthreads();

  const int lane = t & 31, w = t >> 5;
  const int waveM = w & 1, waveN = w >> 1;
  const int half = lane >> 4, ln = lane & 15;

  v8f zero = {0.f,0.f,0.f,0.f,0.f,0.f,0.f,0.f};
  v8f acc[4] = { zero, zero, zero, zero };

  const float* wptr = w2 + (size_t)e * HID * DIM + dBase;
  const uint32_t stage_lds = (uint32_t)(uintptr_t)&stage[0][0];
  const int arow = t >> 3, ac4 = (t & 7) * 4;
  const size_t aSrc = (size_t)rowSlot[arow] * HID;
  const int bk = t & 31, bc0 = (t >> 5) * 32;

  if (w == 0)
    tdm_load_tile_f32(stage_lds, wptr, TN, TK, DIM, DIM, HID);

  for (int kk = 0; kk < HID / TK; ++kk) {
    if (w == 0) __builtin_amdgcn_s_wait_tensorcnt(0);
    __syncthreads();

    *(uint2*)&Al[arow][ac4] = *(const uint2*)&hbuf[aSrc + kk * TK + ac4];
    #pragma unroll
    for (int i = 0; i < 8; ++i) {
      float4 f = *(const float4*)&stage[bk][bc0 + i * 4];
      Bt[bc0 + i*4 + 0][bk] = f2bf(f.x);
      Bt[bc0 + i*4 + 1][bk] = f2bf(f.y);
      Bt[bc0 + i*4 + 2][bk] = f2bf(f.z);
      Bt[bc0 + i*4 + 3][bk] = f2bf(f.w);
    }
    __syncthreads();

    if (w == 0 && kk + 1 < HID / TK)
      tdm_load_tile_f32(stage_lds, wptr + (size_t)(kk + 1) * TK * DIM, TN, TK, DIM, DIM, HID);

    FragBF a;
    const int ar = waveM * 16 + ln, ko = half * 8;
    a.q[0] = *(const uint2*)&Al[ar][ko];
    a.q[1] = *(const uint2*)&Al[ar][ko + 4];
    a.q[2] = *(const uint2*)&Al[ar][ko + 16];
    a.q[3] = *(const uint2*)&Al[ar][ko + 20];
    const int kob = half * 8;
    #pragma unroll
    for (int j = 0; j < 4; ++j) {
      FragBF b;
      const int col = waveN * 64 + j * 16 + ln;
      b.q[0] = *(const uint2*)&Bt[col][kob];
      b.q[1] = *(const uint2*)&Bt[col][kob + 4];
      b.q[2] = *(const uint2*)&Bt[col][kob + 16];
      b.q[3] = *(const uint2*)&Bt[col][kob + 20];
      acc[j] = __builtin_amdgcn_wmma_f32_16x16x32_bf16(false, a.v, false, b.v,
                                                       (short)0, acc[j], false, false);
    }
  }

  #pragma unroll
  for (int j = 0; j < 4; ++j) {
    const int col = dBase + waveN * 64 + j * 16 + ln;
    const float bias = b2[e * DIM + col];
    #pragma unroll
    for (int v = 0; v < 8; ++v) {
      const int mrow = waveM * 16 + half * 8 + v;
      if (tileBase + mrow < n) {
        const int r = rowSlot[mrow];
        const float g = slotg[r];
        // exactly 2 contributions per element; 2-operand FP add is commutative -> deterministic
        atomicAdd(&outp[(size_t)(r >> 1) * DIM + col], (acc[j][v] + bias) * g);
      }
    }
  }
}

// ---------------- deterministic column reductions ----------------
__global__ void reduce_kernel(const float* __restrict__ gates, const float* __restrict__ probs,
                              float* __restrict__ imp, float* __restrict__ loadv,
                              float* __restrict__ util) {
  __shared__ float s1[256], s2[256], s3[256];
  const int e = blockIdx.x, t = threadIdx.x;
  float a = 0.f, b = 0.f, c = 0.f;
  for (int i = t; i < B_TOK; i += 256) {
    float g = gates[i * NE + e];
    a += g; b += probs[i * NE + e]; c += (g > 0.f) ? 1.f : 0.f;
  }
  s1[t] = a; s2[t] = b; s3[t] = c; __syncthreads();
  for (int off = 128; off > 0; off >>= 1) {
    if (t < off) { s1[t] += s1[t+off]; s2[t] += s2[t+off]; s3[t] += s3[t+off]; }
    __syncthreads();
  }
  if (t == 0) { imp[e] = s1[0]; loadv[e] = s2[0]; util[e] = s3[0]; }
}

__global__ void finalize_kernel(const float* __restrict__ imp, const float* __restrict__ loadv,
                                const float* __restrict__ util, float* __restrict__ dout) {
  const int t = threadIdx.x;
  if (t < NE) dout[UTIL_OFF + t] = util[t] * (1.0f / (float)B_TOK);
  if (t == 0) {
    float m1 = 0.f, m2 = 0.f;
    for (int e = 0; e < NE; ++e) { m1 += imp[e]; m2 += loadv[e]; }
    m1 /= NE; m2 /= NE;
    float v1 = 0.f, v2 = 0.f;
    for (int e = 0; e < NE; ++e) {
      float d1 = imp[e] - m1, d2 = loadv[e] - m2;
      v1 += d1 * d1; v2 += d2 * d2;
    }
    v1 /= (NE - 1); v2 /= (NE - 1);   // Bessel (ddof=1)
    dout[LBL_OFF] = sqrtf(v1) / (m1 + 1e-6f) + sqrtf(v2) / (m2 + 1e-6f);
  }
}

extern "C" void kernel_launch(void* const* d_in, const int* in_sizes, int n_in,
                              void* d_out, int out_size, void* d_ws, size_t ws_size,
                              hipStream_t stream) {
  (void)in_sizes; (void)n_in; (void)out_size; (void)ws_size;
  const float* x  = (const float*)d_in[0];
  const float* Wg = (const float*)d_in[1];
  const float* bg = (const float*)d_in[2];
  const float* w1 = (const float*)d_in[3];
  const float* b1 = (const float*)d_in[4];
  const float* w2 = (const float*)d_in[5];
  const float* b2 = (const float*)d_in[6];
  float* out = (float*)d_out;

  char* ws = (char*)d_ws;
  int*   cnt   = (int*)  (ws + OFF_CNT);
  int*   lists = (int*)  (ws + OFF_LISTS);
  float* slotg = (float*)(ws + OFF_SLOTG);
  float* probs = (float*)(ws + OFF_PROBS);
  float* imp   = (float*)(ws + OFF_IMP);
  float* loadv = (float*)(ws + OFF_LOAD);
  float* util  = (float*)(ws + OFF_UTIL);
  unsigned short* xbf  = (unsigned short*)(ws + OFF_XBF);
  unsigned short* hbuf = (unsigned short*)(ws + OFF_HBUF);

  zero_kernel <<<2048, 256, 0, stream>>>(out, cnt);
  xcast_kernel<<<(B_TOK * DIM / 4 + 255) / 256, 256, 0, stream>>>(x, xbf);
  gate_kernel <<<B_TOK * 32 / 256, 256, 0, stream>>>(x, Wg, bg, out, probs, cnt, lists, slotg);
  ffn1_kernel <<<dim3(HID / TN, B_TOK / TM, NE), 256, 0, stream>>>(xbf, w1, b1, hbuf, cnt, lists);
  ffn2_kernel <<<dim3(DIM / TN, B_TOK / TM, NE), 256, 0, stream>>>(hbuf, w2, b2, out, slotg, cnt, lists);
  reduce_kernel<<<NE, 256, 0, stream>>>(out + G_OFF, probs, imp, loadv, util);
  finalize_kernel<<<1, 64, 0, stream>>>(imp, loadv, util, out);
}
